// model1_53953379172890
// MI455X (gfx1250) — matrix operand order repaired and checked by reading, not compiled, not run on hardware
//
#include <hip/hip_runtime.h>
#include <hip/hip_bf16.h>

// Problem dims (fixed by the reference)
#define HH 512   // hidden states
#define DD 128   // tones
#define NN 64    // sequences
#define TT 512   // time steps

typedef float v8f __attribute__((ext_vector_type(8)));
typedef __bf16 v16bf __attribute__((ext_vector_type(16)));
typedef unsigned int v4u __attribute__((ext_vector_type(4)));

union BfPack {
    v4u          q[2];
    unsigned int u[8];
    v16bf        v;
};

__device__ __forceinline__ unsigned short f2bf_rn(float f) {
    unsigned int u = __float_as_uint(f);
    u += 0x7fffu + ((u >> 16) & 1u);
    return (unsigned short)(u >> 16);
}
__device__ __forceinline__ float bf2f(unsigned short h) {
    return __uint_as_float(((unsigned int)h) << 16);
}
// A-operand K base (16-bit A, 16x32 tile, contiguous K = 16*hi + 2j, matching packed B)
__device__ __forceinline__ int a_k0(int j, int hi) {
    return 2 * j + (hi ? 16 : 0);
}

// ---------------------------------------------------------------------------
// Pack probs_x (linear domain!) into WMMA B-operand layout, split bf16 hi/lo.
// Layout: pk[((nt*16 + kc)*32 + lane)*8 + j] holds K = kc*32 + 2j + 16*hi (+half),
// column h = nt*16 + (lane&15).
// ---------------------------------------------------------------------------
__global__ void pack_p_kernel(const float* __restrict__ px,
                              unsigned int* __restrict__ phi,
                              unsigned int* __restrict__ plo) {
    int idx  = blockIdx.x * blockDim.x + threadIdx.x;  // 32*16*32*8 = 131072
    int j    = idx & 7;
    int lane = (idx >> 3) & 31;
    int kc   = (idx >> 8) & 15;
    int nt   = idx >> 12;
    int l16 = lane & 15, hi = lane >> 4;
    int k0 = kc * 32 + 2 * j + (hi ? 16 : 0);
    int h  = nt * 16 + l16;
    float p0 = px[k0 * HH + h];
    float p1 = px[(k0 + 1) * HH + h];
    unsigned short b0 = f2bf_rn(p0), b1 = f2bf_rn(p1);
    phi[idx] = (unsigned)b0 | ((unsigned)b1 << 16);
    plo[idx] = (unsigned)f2bf_rn(p0 - bf2f(b0)) |
               ((unsigned)f2bf_rn(p1 - bf2f(b1)) << 16);
}

// Pack emission weights w[d][h] = log(py[h,d]) - log1p(-py[h,d]) the same way (K=d).
__global__ void pack_w_kernel(const float* __restrict__ py,
                              unsigned int* __restrict__ whi,
                              unsigned int* __restrict__ wlo) {
    int idx  = blockIdx.x * blockDim.x + threadIdx.x;  // 32*4*32*8 = 32768
    int j    = idx & 7;
    int lane = (idx >> 3) & 31;
    int kc   = (idx >> 8) & 3;
    int nt   = idx >> 10;
    int l16 = lane & 15, hi = lane >> 4;
    int d0 = kc * 32 + 2 * j + (hi ? 16 : 0);
    int h  = nt * 16 + l16;
    float q0 = py[h * DD + d0];
    float q1 = py[h * DD + d0 + 1];
    float w0 = __logf(q0) - log1pf(-q0);
    float w1 = __logf(q1) - log1pf(-q1);
    unsigned short b0 = f2bf_rn(w0), b1 = f2bf_rn(w1);
    whi[idx] = (unsigned)b0 | ((unsigned)b1 << 16);
    wlo[idx] = (unsigned)f2bf_rn(w0 - bf2f(b0)) |
               ((unsigned)f2bf_rn(w1 - bf2f(b1)) << 16);
}

// bias[h] = sum_d log1p(-py[h,d]);  lp0[h] = log(probs_x[0,h])
__global__ void bias_lp0_kernel(const float* __restrict__ px,
                                const float* __restrict__ py,
                                float* __restrict__ bias,
                                float* __restrict__ lp0) {
    int h = blockIdx.x * blockDim.x + threadIdx.x;
    if (h >= HH) return;
    float s = 0.f;
    for (int d = 0; d < DD; ++d) s += log1pf(-py[h * DD + d]);
    bias[h] = s;
    lp0[h]  = __logf(px[h]);
}

// ---------------------------------------------------------------------------
// Emission GEMM: emit[m, h] = seq[m,:]·w[:,h] + bias[h], m = n*T + t.
// A = sequences (0/1 -> exact in bf16), B = w split hi/lo -> 2 WMMAs per K-chunk.
// One 16x16 tile per wave, 8 waves/block, 8192 blocks (65536 tiles).
// ---------------------------------------------------------------------------
__global__ void __launch_bounds__(256)
emit_kernel(const float* __restrict__ seq,
            const unsigned int* __restrict__ whi,
            const unsigned int* __restrict__ wlo,
            const float* __restrict__ bias,
            float* __restrict__ emit) {
    const int lane = threadIdx.x & 31;
    const int wave = threadIdx.x >> 5;
    const int tile = blockIdx.x * 8 + wave;
    const int mt = tile >> 5;   // row tile (0..2047)
    const int nt = tile & 31;   // h tile
    const int m0 = mt * 16, h0t = nt * 16;
    const int l16 = lane & 15, hi = lane >> 4;

    v8f acc = {0.f, 0.f, 0.f, 0.f, 0.f, 0.f, 0.f, 0.f};
#pragma unroll
    for (int kc = 0; kc < 4; ++kc) {
        BfPack A, Bh, Bl;
#pragma unroll
        for (int j = 0; j < 8; ++j) {
            int k0 = a_k0(j, hi);
            const float* sp = seq + (m0 + l16) * DD + kc * 32 + k0;
            float s0 = sp[0], s1 = sp[1];
            // truncation is exact for 0.0f / 1.0f
            A.u[j] = (__float_as_uint(s0) >> 16) | (__float_as_uint(s1) & 0xffff0000u);
        }
        const v4u* bh = (const v4u*)(whi + ((nt * 4 + kc) * 32 + lane) * 8);
        const v4u* bl = (const v4u*)(wlo + ((nt * 4 + kc) * 32 + lane) * 8);
        Bh.q[0] = bh[0]; Bh.q[1] = bh[1];
        Bl.q[0] = bl[0]; Bl.q[1] = bl[1];
        acc = __builtin_amdgcn_wmma_f32_16x16x32_bf16(false, A.v, false, Bh.v,
                                                      (short)0, acc, false, false);
        acc = __builtin_amdgcn_wmma_f32_16x16x32_bf16(false, A.v, false, Bl.v,
                                                      (short)0, acc, false, false);
    }
    float b = bias[h0t + l16];
#pragma unroll
    for (int r = 0; r < 8; ++r) {
        // C/D layout: VGPR r <-> row M = r (+8 for lanes 16..31), col = lane&15
        emit[(m0 + r + (hi ? 8 : 0)) * HH + h0t + l16] = acc[r] + b;
    }
}

// ---------------------------------------------------------------------------
// Forward scan. One workgroup (32 waves, 1024 thr) per 16 sequences; alpha lives
// in LDS. Per step: row-max -> exp rescale (bf16 hi/lo) -> [16x512]x[512x512]
// bf16x3 WMMA GEMM against packed P streamed from L2 -> log + emit + length mask.
// Ends with per-row logsumexp written to d_out.
// ---------------------------------------------------------------------------
#define ASTR 516  // alpha row stride (floats): conflict-free writeback
#define PSTR 261  // a_hi/a_lo row stride (uint pairs): conflict-free A gathers

__global__ void __launch_bounds__(1024)
scan_kernel(const float* __restrict__ emit,
            const unsigned int* __restrict__ phi,
            const unsigned int* __restrict__ plo,
            const float* __restrict__ lp0,
            const int* __restrict__ lengths,
            float* __restrict__ out) {
    __shared__ float        alpha[16 * ASTR];
    __shared__ unsigned int ahi[16 * PSTR];
    __shared__ unsigned int alo[16 * PSTR];
    __shared__ float        mrow[16];
    __shared__ int          lenr[16];

    const int tid  = threadIdx.x;
    const int lane = tid & 31;
    const int wave = tid >> 5;           // 0..31, owns h-tile h0 = wave*16
    const int l16 = lane & 15, hi = lane >> 4;
    const int n0 = blockIdx.x * 16;
    const int h0t = wave * 16;

    if (tid < 16) lenr[tid] = lengths[n0 + tid];

    // t = 0: alpha[r][h] = log(px[0,h]) + emit[n0+r, 0, h]
    for (int i = tid; i < 16 * HH; i += 1024) {
        int r = i >> 9, h = i & (HH - 1);
        alpha[r * ASTR + h] = lp0[h] + emit[(size_t)(n0 + r) * TT * HH + h];
    }
    __syncthreads();

    for (int t = 1; t < TT; ++t) {
        // phase 1: row max (first 16 waves, one row each)
        if (wave < 16) {
            const int r = wave;
            float m = -3.4e38f;
#pragma unroll
            for (int i = 0; i < 16; ++i)
                m = fmaxf(m, alpha[r * ASTR + lane + 32 * i]);
#pragma unroll
            for (int s = 16; s > 0; s >>= 1)
                m = fmaxf(m, __shfl_xor(m, s, 32));
            if (lane == 0) mrow[r] = m;
        }
        __syncthreads();
        // phase 2: a = exp(alpha - m) split into bf16 hi/lo pairs
        for (int p = tid; p < 16 * 256; p += 1024) {
            int r = p >> 8, kp = p & 255;
            float m  = mrow[r];
            float e0 = __expf(alpha[r * ASTR + 2 * kp] - m);
            float e1 = __expf(alpha[r * ASTR + 2 * kp + 1] - m);
            unsigned short b0 = f2bf_rn(e0), b1 = f2bf_rn(e1);
            ahi[r * PSTR + kp] = (unsigned)b0 | ((unsigned)b1 << 16);
            alo[r * PSTR + kp] = (unsigned)f2bf_rn(e0 - bf2f(b0)) |
                                 ((unsigned)f2bf_rn(e1 - bf2f(b1)) << 16);
        }
        __syncthreads();
        // phase 3: acc = a @ P for this wave's 16-wide h' tile (bf16x3)
        v8f acc = {0.f, 0.f, 0.f, 0.f, 0.f, 0.f, 0.f, 0.f};
        const unsigned int* pb_hi = phi + (size_t)wave * 4096;  // nt = wave
        const unsigned int* pb_lo = plo + (size_t)wave * 4096;
#pragma unroll 4
        for (int kc = 0; kc < 16; ++kc) {
            BfPack Ah, Al, Bh, Bl;
#pragma unroll
            for (int j = 0; j < 8; ++j) {
                int kp = kc * 16 + (a_k0(j, hi) >> 1);
                Ah.u[j] = ahi[l16 * PSTR + kp];
                Al.u[j] = alo[l16 * PSTR + kp];
            }
            const v4u* bh = (const v4u*)(pb_hi + (kc * 32 + lane) * 8);
            const v4u* bl = (const v4u*)(pb_lo + (kc * 32 + lane) * 8);
            Bh.q[0] = bh[0]; Bh.q[1] = bh[1];
            Bl.q[0] = bl[0]; Bl.q[1] = bl[1];
            if (kc + 1 < 16) {  // global_prefetch_b8 of next K-chunk
                __builtin_prefetch(pb_hi + ((kc + 1) * 32 + lane) * 8, 0, 0);
                __builtin_prefetch(pb_lo + ((kc + 1) * 32 + lane) * 8, 0, 0);
            }
            acc = __builtin_amdgcn_wmma_f32_16x16x32_bf16(false, Ah.v, false, Bh.v,
                                                          (short)0, acc, false, false);
            acc = __builtin_amdgcn_wmma_f32_16x16x32_bf16(false, Ah.v, false, Bl.v,
                                                          (short)0, acc, false, false);
            acc = __builtin_amdgcn_wmma_f32_16x16x32_bf16(false, Al.v, false, Bh.v,
                                                          (short)0, acc, false, false);
        }
        // phase 4: back to log domain + emission + length mask, write alpha
#pragma unroll
        for (int r8 = 0; r8 < 8; ++r8) {
            int r = r8 + (hi ? 8 : 0);
            float old = alpha[r * ASTR + h0t + l16];
            float nl = __logf(acc[r8]) + mrow[r] +
                       emit[((size_t)(n0 + r) * TT + t) * HH + h0t + l16];
            alpha[r * ASTR + h0t + l16] = (t < lenr[r]) ? nl : old;
        }
        __syncthreads();
    }

    // final per-row logsumexp -> out[n]
    if (wave < 16) {
        const int r = wave;
        float m = -3.4e38f;
#pragma unroll
        for (int i = 0; i < 16; ++i)
            m = fmaxf(m, alpha[r * ASTR + lane + 32 * i]);
#pragma unroll
        for (int s = 16; s > 0; s >>= 1)
            m = fmaxf(m, __shfl_xor(m, s, 32));
        float ssum = 0.f;
#pragma unroll
        for (int i = 0; i < 16; ++i)
            ssum += __expf(alpha[r * ASTR + lane + 32 * i] - m);
#pragma unroll
        for (int s = 16; s > 0; s >>= 1)
            ssum += __shfl_xor(ssum, s, 32);
        if (lane == 0) out[n0 + r] = m + __logf(ssum);
    }
}

extern "C" void kernel_launch(void* const* d_in, const int* in_sizes, int n_in,
                              void* d_out, int out_size, void* d_ws, size_t ws_size,
                              hipStream_t stream) {
    (void)in_sizes; (void)n_in; (void)out_size; (void)ws_size;
    const float* seq     = (const float*)d_in[0];   // [N,T,D] f32
    const int*   lengths = (const int*)d_in[1];     // [N] i32
    const float* px      = (const float*)d_in[2];   // [H,H] f32
    const float* py      = (const float*)d_in[3];   // [H,D] f32
    float* out = (float*)d_out;                     // [N] f32

    char* ws = (char*)d_ws;
    size_t off = 0;
    float* emit = (float*)(ws + off);        off += (size_t)NN * TT * HH * 4;  // 64 MB
    unsigned int* phi = (unsigned int*)(ws + off); off += (size_t)32 * 16 * 32 * 8 * 4;
    unsigned int* plo = (unsigned int*)(ws + off); off += (size_t)32 * 16 * 32 * 8 * 4;
    unsigned int* whi = (unsigned int*)(ws + off); off += (size_t)32 * 4 * 32 * 8 * 4;
    unsigned int* wlo = (unsigned int*)(ws + off); off += (size_t)32 * 4 * 32 * 8 * 4;
    float* bias = (float*)(ws + off);        off += HH * 4;
    float* lp0  = (float*)(ws + off);        off += HH * 4;

    pack_p_kernel<<<512, 256, 0, stream>>>(px, phi, plo);
    pack_w_kernel<<<128, 256, 0, stream>>>(py, whi, wlo);
    bias_lp0_kernel<<<2, 256, 0, stream>>>(px, py, bias, lp0);
    emit_kernel<<<8192, 256, 0, stream>>>(seq, whi, wlo, bias, emit);
    scan_kernel<<<4, 1024, 0, stream>>>(emit, phi, plo, lp0, lengths, out);
}
